// RealSymplecticReactionDiffusion2D_89730456748911
// MI455X (gfx1250) — compile-verified, matrix operand order checked by reading, not run on hardware
//
#include <hip/hip_runtime.h>

typedef __attribute__((ext_vector_type(16))) _Float16 v16h;
typedef __attribute__((ext_vector_type(8)))  _Float16 v8h;
typedef __attribute__((ext_vector_type(8)))  float    v8f;

#define B_   8
#define N_   4096
#define D_   768
#define K2_  1536
#define HW_  64

__device__ __forceinline__ float softplusf(float z) {
  return z > 20.f ? z : log1pf(expf(z));
}

// ---------------- pooled mean (two-stage deterministic reduction) ------------
__global__ void k_pool1(const float* __restrict__ x, float* __restrict__ part) {
  int d = blockIdx.x * 256 + threadIdx.x;      // 0..767
  int chunk = blockIdx.y;                      // 0..31 (128 rows each)
  int b = blockIdx.z;
  const float* p = x + ((size_t)b * N_ + (size_t)chunk * 128) * D_ + d;
  float s = 0.f;
  for (int n = 0; n < 128; ++n) s += p[(size_t)n * D_];
  part[((size_t)b * 32 + chunk) * D_ + d] = s;
}

__global__ void k_pool2(const float* __restrict__ part, float* __restrict__ pooled) {
  int d = blockIdx.x * 256 + threadIdx.x;
  int b = blockIdx.y;
  float s = 0.f;
  for (int c = 0; c < 32; ++c) s += part[((size_t)b * 32 + c) * D_ + d];
  pooled[b * D_ + d] = s * (1.0f / N_);
}

// ---------------- gate: gamma = min(softplus(pooled @ Wg^T + bg), 0.05) ------
__global__ void k_gamma(const float* __restrict__ pooled, const float* __restrict__ Wg,
                        const float* __restrict__ bg, float* __restrict__ gamma) {
  __shared__ float sp[D_];
  int b = blockIdx.y;
  for (int i = threadIdx.x; i < D_; i += 256) sp[i] = pooled[b * D_ + i];
  __syncthreads();
  int j = blockIdx.x * 256 + threadIdx.x;
  float acc = bg[j];
  const float* w = Wg + (size_t)j * D_;
  for (int k = 0; k < D_; ++k) acc += sp[k] * w[k];
  gamma[b * D_ + j] = fminf(softplusf(acc), 0.05f);
}

// ---------------- W_out [768,1536] f32 -> PB: fragment-ready f16 B tiles -----
// PB[kc][jt][lane][16] = B[k = kc*32 + kmap(lane,i)][j = jt*16 + lane%16]
// kmap matches the wave32 16-bit B-matrix VGPR layout (lanes>=16 hold K+8).
__global__ void k_woutP(const float* __restrict__ Wout, _Float16* __restrict__ PB) {
  int kc = blockIdx.x;           // 0..47
  int jt = blockIdx.y;           // 0..47
  int lane = threadIdx.x;        // 0..31
  int lh = lane >> 4, ll = lane & 15;
  int j = jt * 16 + ll;
  _Float16 v[16];
#pragma unroll
  for (int i = 0; i < 16; ++i) {
    int k = kc * 32 + (i < 8 ? i + 8 * lh : 16 + (i - 8) + 8 * lh);
    v[i] = (_Float16)Wout[(size_t)j * K2_ + k];
  }
  _Float16* dst = PB + (((size_t)kc * 48 + jt) * 32 + lane) * 16;
  *(uint4*)dst = *(uint4*)v;
  *(uint4*)(dst + 8) = *(uint4*)(v + 8);
}

// ---------------- x [B,N,d] -> uT [B,d,N] (tiled transpose) ------------------
__global__ void k_xT(const float* __restrict__ x, float* __restrict__ uT) {
  __shared__ float T[32][33];
  int d0 = blockIdx.x * 32, n0 = blockIdx.y * 32, b = blockIdx.z;
  int tx = threadIdx.x, ty = threadIdx.y;
  for (int q = 0; q < 4; ++q)
    T[ty + 8 * q][tx] = x[((size_t)b * N_ + n0 + ty + 8 * q) * D_ + d0 + tx];
  __syncthreads();
  for (int q = 0; q < 4; ++q)
    uT[((size_t)b * D_ + d0 + ty + 8 * q) * N_ + n0 + tx] = T[tx][ty + 8 * q];
}

// ---------------- symplectic PDE: one 64x64 plane per workgroup in LDS -------
__device__ __forceinline__ float lapf(const float* z, int i) {
  int h = i >> 6, w = i & 63;
  int wl = (w - 1) & 63, wr = (w + 1) & 63;
  int hu = (h - 1) & 63, hd = (h + 1) & 63;
  return z[(h << 6) | wl] + z[(h << 6) | wr] + z[(hu << 6) | w] + z[(hd << 6) | w] - 4.f * z[i];
}

__global__ void __launch_bounds__(256) k_pde(const float* __restrict__ uT,
                                             const float* __restrict__ gammaB,
                                             const float* __restrict__ alpha,
                                             const float* __restrict__ beta,
                                             const int* __restrict__ kptr,
                                             _Float16* __restrict__ comb) {
  __shared__ float su[N_];
  __shared__ float sv[N_];
  int d = blockIdx.x, b = blockIdx.y, t = threadIdx.x;
  const float* g0 = uT + ((size_t)b * D_ + d) * N_;
  for (int j = 0; j < 16; ++j) { int i = t + j * 256; su[i] = g0[i]; sv[i] = 0.f; }
  int ks = kptr[0]; if (ks < 1) ks = 1;
  float dt = 1.0f / (float)ks, hdt = 0.5f * dt;
  float g  = gammaB[b * D_ + d];
  float ae = 0.2f * tanhf(alpha[d]);
  float be = fmaxf(softplusf(beta[d]), 1e-4f);
  float c1 = hdt * g, c2 = dt * g;
  int base = t * 16;
  __syncthreads();
  for (int s = 0; s < ks; ++s) {
    for (int e = 0; e < 16; ++e) {                 // reaction half-step
      int i = base + e; float u = su[i], v = sv[i];
      float r = fminf(1.f, fmaxf(-1.f, ae - be * (u * u + v * v)));
      su[i] = u + hdt * r * u; sv[i] = v + hdt * r * v;
    }
    __syncthreads();
    for (int e = 0; e < 16; ++e) { int i = base + e; sv[i] -= c1 * lapf(su, i); }  // v_half
    __syncthreads();
    for (int e = 0; e < 16; ++e) { int i = base + e; su[i] += c2 * lapf(sv, i); }  // u_new
    __syncthreads();
    for (int e = 0; e < 16; ++e) { int i = base + e; sv[i] -= c1 * lapf(su, i); }  // v_new
    __syncthreads();
    for (int e = 0; e < 16; ++e) {                 // reaction half-step
      int i = base + e; float u = su[i], v = sv[i];
      float r = fminf(1.f, fmaxf(-1.f, ae - be * (u * u + v * v)));
      su[i] = u + hdt * r * u; sv[i] = v + hdt * r * v;
    }
    __syncthreads();
  }
  _Float16* cu = comb + ((size_t)b * K2_ + d) * N_;
  _Float16* cv = comb + ((size_t)b * K2_ + D_ + d) * N_;
  for (int j = 0; j < 16; ++j) {
    int i = t + j * 256;
    cu[i] = (_Float16)su[i];
    cv[i] = (_Float16)sv[i];
  }
}

// ---------------- comb [B,K2,N] f16 -> combN [B,N,K2] f16 (tiled transpose) --
__global__ void k_combT(const _Float16* __restrict__ comb, _Float16* __restrict__ combN) {
  __shared__ _Float16 T[64][66];   // [n][k], padded
  int k0 = blockIdx.x * 64, n0 = blockIdx.y * 64, b = blockIdx.z;
  int tx = threadIdx.x, ty = threadIdx.y;   // 32 x 8
  for (int q = 0; q < 8; ++q) {
    int kr = ty + 8 * q;
    uint v = *(const uint*)(comb + ((size_t)b * K2_ + k0 + kr) * N_ + n0 + tx * 2);
    T[tx * 2][kr]     = ((const _Float16*)&v)[0];
    T[tx * 2 + 1][kr] = ((const _Float16*)&v)[1];
  }
  __syncthreads();
  for (int q = 0; q < 8; ++q) {
    int nr = ty + 8 * q;
    _Float16 two[2] = { T[nr][tx * 2], T[nr][tx * 2 + 1] };
    *(uint*)(combN + ((size_t)b * N_ + n0 + nr) * K2_ + k0 + tx * 2) = *(uint*)two;
  }
}

// ---------------- WMMA GEMM: out = combined @ Wout^T + x*D -------------------
// A[m,k] = combN[b,m,k] (f16 row-major), B from fragment-packed PB (L2-resident).
// As rows are 40 halves (80B) so every b128 LDS access is 16B aligned.
__device__ __forceinline__ v16h ldfrag(const _Float16 (*S)[40], int row, int half) {
  v8h lo = *(const v8h*)&S[row][8 * half];        // k = 8*half .. +7
  v8h hi = *(const v8h*)&S[row][16 + 8 * half];   // k = 16+8*half .. +7
  v16h r;
#pragma unroll
  for (int i = 0; i < 8; ++i) { r[i] = lo[i]; r[i + 8] = hi[i]; }
  return r;
}

__device__ __forceinline__ v16h ldPB(const _Float16* __restrict__ PB, int kc, int jt, int lane) {
  const uint4* p = (const uint4*)(PB + (((size_t)kc * 48 + jt) * 32 + lane) * 16);
  union { uint4 u[2]; v16h h; } cvt;
  cvt.u[0] = p[0];
  cvt.u[1] = p[1];
  return cvt.h;
}

__global__ void __launch_bounds__(128) k_gemm(const _Float16* __restrict__ combN,
                                              const _Float16* __restrict__ PB,
                                              const float* __restrict__ x,
                                              const float* __restrict__ Dc,
                                              float* __restrict__ out) {
  __shared__ __align__(16) _Float16 As[2][64][40];   // double-buffered [m_local][k_local]
  int t = threadIdx.x;
  int m0 = blockIdx.x * 64, j0 = blockIdx.y * 64, bb = blockIdx.z;
  int wave = t >> 5, lane = t & 31;
  int wm = wave >> 1, wn = wave & 1;                 // 2x2 waves, 32x32 each
  int lhalf = lane >> 4, llo = lane & 15;
  int jt0 = blockIdx.y * 4 + wn * 2;                 // wave's first 16-wide j tile
  v8f acc00 = {}, acc01 = {}, acc10 = {}, acc11 = {};
  int row = t >> 1, part = (t & 1) << 4;             // stage: 64 rows x 2 half-rows
  const _Float16* gA = combN + ((size_t)bb * N_ + m0 + row) * K2_ + part;
  for (int k0 = 0; k0 < K2_; k0 += 32) {
    int buf = (k0 >> 5) & 1;
    int kc = k0 >> 5;
    uint4 a0 = *(const uint4*)(gA);                  // 16 halves of A row
    uint4 a1 = *(const uint4*)(gA + 8);
    *(uint4*)&As[buf][row][part]     = a0;           // ds_store_b128 (16B aligned)
    *(uint4*)&As[buf][row][part + 8] = a1;
    v16h bF0 = ldPB(PB, kc, jt0, lane);              // global b128 x2, L2-resident
    v16h bF1 = ldPB(PB, kc, jt0 + 1, lane);
    __syncthreads();
    v16h aF0 = ldfrag(As[buf], wm * 32 + llo, lhalf);
    v16h aF1 = ldfrag(As[buf], wm * 32 + 16 + llo, lhalf);
    acc00 = __builtin_amdgcn_wmma_f32_16x16x32_f16(false, aF0, false, bF0, (short)0, acc00, false, false);
    acc01 = __builtin_amdgcn_wmma_f32_16x16x32_f16(false, aF0, false, bF1, (short)0, acc01, false, false);
    acc10 = __builtin_amdgcn_wmma_f32_16x16x32_f16(false, aF1, false, bF0, (short)0, acc10, false, false);
    acc11 = __builtin_amdgcn_wmma_f32_16x16x32_f16(false, aF1, false, bF1, (short)0, acc11, false, false);
    gA += 32;
  }
  // epilogue: D-layout — VGPR r: lanes0-15 -> M=r, lanes16-31 -> M=r+8; N=lane%16
#pragma unroll
  for (int tm = 0; tm < 2; ++tm) {
#pragma unroll
    for (int tn = 0; tn < 2; ++tn) {
      v8f a = tm ? (tn ? acc11 : acc10) : (tn ? acc01 : acc00);
#pragma unroll
      for (int r = 0; r < 8; ++r) {
        int gm = m0 + wm * 32 + tm * 16 + r + 8 * lhalf;
        int gj = j0 + wn * 32 + tn * 16 + llo;
        size_t o = ((size_t)bb * N_ + gm) * D_ + gj;
        out[o] = a[r] + x[o] * Dc[gj];
      }
    }
  }
}

extern "C" void kernel_launch(void* const* d_in, const int* in_sizes, int n_in,
                              void* d_out, int out_size, void* d_ws, size_t ws_size,
                              hipStream_t stream) {
  const float* x     = (const float*)d_in[0];
  const float* Wg    = (const float*)d_in[1];
  const float* bg    = (const float*)d_in[2];
  const float* alpha = (const float*)d_in[3];
  const float* beta  = (const float*)d_in[4];
  const float* Wout  = (const float*)d_in[5];
  const float* Dc    = (const float*)d_in[6];
  const int*   kptr  = (const int*)d_in[7];
  float* out = (float*)d_out;

  char* ws = (char*)d_ws;
  size_t off = 0;
  float*    part   = (float*)(ws + off);    off += (size_t)B_ * 32 * D_ * 4;
  float*    pooled = (float*)(ws + off);    off += (size_t)B_ * D_ * 4;
  float*    gamma  = (float*)(ws + off);    off += (size_t)B_ * D_ * 4;
  _Float16* PB     = (_Float16*)(ws + off); off += (size_t)48 * 48 * 32 * 16 * 2;
  off = (off + 255) & ~(size_t)255;
  // uT (f32, [B,d,N]) is dead after k_pde; combN (f16, [B,N,K2]) reuses its space.
  char* shared_region = ws + off;           off += (size_t)B_ * D_ * N_ * 4;  // == B*N*K2*2
  float*    uT    = (float*)shared_region;
  _Float16* combN = (_Float16*)shared_region;
  off = (off + 255) & ~(size_t)255;
  _Float16* comb   = (_Float16*)(ws + off); off += (size_t)B_ * K2_ * N_ * 2;
  if (ws_size < off) return;

  k_pool1<<<dim3(3, 32, 8), 256, 0, stream>>>(x, part);
  k_pool2<<<dim3(3, 8), 256, 0, stream>>>(part, pooled);
  k_gamma<<<dim3(3, 8), 256, 0, stream>>>(pooled, Wg, bg, gamma);
  k_woutP<<<dim3(48, 48), 32, 0, stream>>>(Wout, PB);
  k_xT<<<dim3(24, 128, 8), dim3(32, 8), 0, stream>>>(x, uT);
  k_pde<<<dim3(768, 8), 256, 0, stream>>>(uT, gamma, alpha, beta, kptr, comb);
  k_combT<<<dim3(24, 64, 8), dim3(32, 8), 0, stream>>>(comb, combN);
  k_gemm<<<dim3(64, 12, 8), 128, 0, stream>>>(combN, PB, x, Dc, out);
}
